// BlockConv_10514079941182
// MI455X (gfx1250) — matrix operand order repaired and checked by reading, not compiled, not run on hardware
//
#include <hip/hip_runtime.h>
#include <stdint.h>

typedef __attribute__((ext_vector_type(16))) __bf16 v16bf;
typedef __attribute__((ext_vector_type(8)))  float  v8f;

#define BATCH 32
#define CIN   128
#define COUT  128
#define NPIX  128
#define CHUNK 32            // c_in per WMMA K-slab
#define NCHUNK (CIN / CHUNK)
#define XHALO (NPIX + 2)    // 130: x0-1 .. x0+128

// fp32 -> bf16, round-to-nearest-even
__device__ __forceinline__ uint16_t f32_to_bf16(float f) {
    uint32_t u = __float_as_uint(f);
    u += 0x7FFFu + ((u >> 16) & 1u);
    return (uint16_t)(u >> 16);
}

__global__ __launch_bounds__(256)
void blockconv_wmma_bf16(const float* __restrict__ x,
                         const float* __restrict__ w,
                         float* __restrict__ out)
{
    // weights for one ky row: [kx 0..2][o 0..127][c 0..31] bf16  (24576 B)
    __shared__ __attribute__((aligned(16))) uint16_t lds_w[3 * COUT * CHUNK];
    // input rows y-1..y+1 with x halo: [r 0..2][xi 0..129][c 0..31] bf16 (24960 B)
    __shared__ __attribute__((aligned(16))) uint16_t lds_x[3 * XHALO * CHUNK];

    const int tid  = threadIdx.x;
    const int bid  = blockIdx.x;
    const int y    = bid & (NPIX - 1);
    const int b    = bid >> 7;

    const int wave = tid >> 5;
    const int lane = tid & 31;
    const int h    = lane >> 4;   // lane half (0/1)
    const int ln   = lane & 15;

    const int cout_base = (wave >> 1) * 32;  // 4 cout groups of 32
    const int px_base   = (wave & 1) * 64;   // 2 pixel groups of 64

    v8f acc[2][4];
#pragma unroll
    for (int i = 0; i < 2; ++i)
#pragma unroll
        for (int j = 0; j < 4; ++j)
            acc[i][j] = v8f{};

    for (int cc = 0; cc < NCHUNK; ++cc) {
        const int cbase = cc * CHUNK;

        __syncthreads();  // previous chunk fully consumed before restaging input

        // ---- stage input tile: rows y-1..y+1, x = -1..128, 32 channels ----
        for (int j = tid; j < 3 * CHUNK * XHALO; j += 256) {
            int row = j / XHALO;          // row = r*CHUNK + c
            int xi  = j - row * XHALO;    // coalesced along x
            int r   = row / CHUNK;
            int c   = row - r * CHUNK;
            int gy  = y + r - 1;
            int gx  = xi - 1;
            float v = 0.0f;
            if ((unsigned)gy < (unsigned)NPIX && (unsigned)gx < (unsigned)NPIX)
                v = x[((size_t)(b * CIN + cbase + c) * NPIX + gy) * NPIX + gx];
            lds_x[(r * XHALO + xi) * CHUNK + c] = f32_to_bf16(v);
        }

        // pull next weight chunk toward L2 while we compute this one
        if (cc + 1 < NCHUNK)
            __builtin_prefetch(w + (size_t)(cbase + CHUNK) * 9 + (size_t)tid * 288, 0, 1);

        for (int ky = 0; ky < 3; ++ky) {
            __syncthreads();  // previous ky's weights consumed; input tile visible

            // ---- stage weights for this ky: [kx][o][c] ----
            for (int e = tid; e < 3 * COUT * CHUNK; e += 256) {
                int o   = e / (CHUNK * 3);
                int rem = e - o * (CHUNK * 3);
                int c   = rem / 3;
                int kx  = rem - c * 3;
                float v = w[(size_t)o * (CIN * 9) + (size_t)(cbase + c) * 9 + ky * 3 + kx];
                lds_w[(kx * COUT + o) * CHUNK + c] = f32_to_bf16(v);
            }
            __syncthreads();

#pragma unroll
            for (int kx = 0; kx < 3; ++kx) {
                // A tiles: 16 cout x 32 cin. Lane half h: VGPR0-3 = K 8h..8h+7,
                // VGPR4-7 = K 16+8h..16+8h+7  (ISA 7.12.2, 16-bit A 16x32)
                union { uint4 q[2]; v16bf v; } A[2];
#pragma unroll
                for (int ct = 0; ct < 2; ++ct) {
                    int o = cout_base + ct * 16 + ln;
                    const uint16_t* base = &lds_w[(kx * COUT + o) * CHUNK];
                    A[ct].q[0] = *(const uint4*)(base + 8 * h);
                    A[ct].q[1] = *(const uint4*)(base + 16 + 8 * h);
                }
                // B tiles: 32 cin x 16 px. Lane = column px, lane half h holds
                // K = 8h..8h+7, then 16+8h..16+8h+7 (same order as A)
                union { uint4 q[2]; v16bf v; } Bt[4];
#pragma unroll
                for (int pt = 0; pt < 4; ++pt) {
                    int px = px_base + pt * 16 + ln;
                    int xi = px + kx;                 // gx = px + kx - 1, halo index
                    const uint16_t* base = &lds_x[(ky * XHALO + xi) * CHUNK + 8 * h];
                    Bt[pt].q[0] = *(const uint4*)(base);
                    Bt[pt].q[1] = *(const uint4*)(base + 16);
                }
#pragma unroll
                for (int ct = 0; ct < 2; ++ct)
#pragma unroll
                    for (int pt = 0; pt < 4; ++pt)
                        acc[ct][pt] = __builtin_amdgcn_wmma_f32_16x16x32_bf16(
                            false, A[ct].v, false, Bt[pt].v,
                            (short)0, acc[ct][pt], false, false);
            }
        }
    }

    // ---- epilogue: C/D layout — VGPR r: M = r + 8h, N = lane&15 ----
#pragma unroll
    for (int ct = 0; ct < 2; ++ct) {
#pragma unroll
        for (int pt = 0; pt < 4; ++pt) {
            int px = px_base + pt * 16 + ln;
#pragma unroll
            for (int r = 0; r < 8; ++r) {
                int o = cout_base + ct * 16 + 8 * h + r;
                out[((size_t)(b * COUT + o) * NPIX + y) * NPIX + px] = acc[ct][pt][r];
            }
        }
    }
}

extern "C" void kernel_launch(void* const* d_in, const int* in_sizes, int n_in,
                              void* d_out, int out_size, void* d_ws, size_t ws_size,
                              hipStream_t stream) {
    const float* x = (const float*)d_in[0];   // [32,128,128,128] fp32
    const float* w = (const float*)d_in[1];   // [128,128,3,3]   fp32
    float* out = (float*)d_out;               // [32,128,128,128] fp32
    (void)in_sizes; (void)n_in; (void)out_size; (void)d_ws; (void)ws_size;

    dim3 grid(BATCH * NPIX);   // 4096 blocks: one (b, y) image row each
    dim3 block(256);           // 8 wave32
    blockconv_wmma_bf16<<<grid, block, 0, stream>>>(x, w, out);
}
